// AbstractAtt_49031346651584
// MI455X (gfx1250) — matrix operand order repaired and checked by reading, not compiled, not run on hardware
//
#include <hip/hip_runtime.h>
#include <cstdint>
#include <cstddef>

// ---------------------------------------------------------------------------
// Types for CDNA5 WMMA (gfx1250, wave32)
// ---------------------------------------------------------------------------
typedef float        v8f    __attribute__((ext_vector_type(8)));
typedef __bf16       v16bf  __attribute__((ext_vector_type(16)));
typedef unsigned int uint4v __attribute__((ext_vector_type(4)));

union ABu {
  v16bf        v;
  unsigned int u[8];
  uint4v       q[2];
};

#define B_    128
#define DV_   2048    // also DQ and DH
#define S_    196
#define DA_   1200
#define G_    4
#define DH_   2048
#define NANS_ 3000
#define KT_   64      // 2048 / 32 k-tiles
#define NT_   75      // 1200 / 16 n-tiles (attention dim)
#define MROWS_TOTAL (B_ * S_)   // 25088 = 392 * 64

__device__ __forceinline__ unsigned short f2bf(float f) {
  unsigned int u = __float_as_uint(f);
  u += 0x7FFFu + ((u >> 16) & 1u);          // round-to-nearest-even
  return (unsigned short)(u >> 16);
}

// Branchless tanh: 1 - 2/(exp(2x)+1).  v_exp_f32 + v_rcp_f32, no exec-mask
// branches (libm tanhf expands to a divergent multi-path sequence in the hot
// loop).  Saturates correctly: exp->inf gives +1, exp->0 gives -1.
__device__ __forceinline__ float fast_tanh(float x) {
  const float e = __expf(2.0f * x);
  return 1.0f - 2.0f * __frcp_rn(e + 1.0f);
}

__device__ __forceinline__ v8f wmma_bf16(ABu a, ABu b, v8f c) {
  return __builtin_amdgcn_wmma_f32_16x16x32_bf16(
      false, a.v, false, b.v, (short)0, c, false, false);
}

// ---------------------------------------------------------------------------
// Swizzle W[K,N] (f32, row stride rowStride) -> bf16 tiles in per-lane WMMA
// B-operand order: tile = [nt][kt][lane][8 x u32].
// Lane mapping (B): n = nt*16 + (lane&15); khalf = lane>>4;
//   u32 j holds K = kt*32 + (j&3)*2 + khalf*8 + (j>>2)*16 (low) and +1 (high).
// ---------------------------------------------------------------------------
__global__ void conv_swz_w(const float* __restrict__ W, unsigned int* __restrict__ dst,
                           int N, int rowStride) {
  const int kt   = blockIdx.x;          // 0..63
  const int nt   = blockIdx.y;
  const int lane = threadIdx.x;         // 0..31
  const int n     = nt * 16 + (lane & 15);
  const int khalf = lane >> 4;
  unsigned int* o = dst + (((size_t)nt * KT_ + kt) * 32 + lane) * 8;
#pragma unroll
  for (int j = 0; j < 8; ++j) {
    const int k = kt * 32 + (j & 3) * 2 + khalf * 8 + (j >> 2) * 16;
    float f0 = 0.f, f1 = 0.f;
    if (n < N) {
      f0 = W[(size_t)k * rowStride + n];
      f1 = W[(size_t)(k + 1) * rowStride + n];
    }
    o[j] = (unsigned int)f2bf(f0) | ((unsigned int)f2bf(f1) << 16);
  }
}

// ---------------------------------------------------------------------------
// Swizzle X[M, K=2048] (f32, row stride rowStride) -> bf16 A tiles
// A-operand order per ISA table: m = mt*16 + (lane&15); khalf = lane>>4;
//   u32 j holds K = kt*32 + (j&3)*2 + khalf*8 + (j>>2)*16 (low) and +1 (high).
// ---------------------------------------------------------------------------
__global__ void conv_swz_a(const float* __restrict__ X, unsigned int* __restrict__ dst,
                           int rowStride) {
  const int kt   = blockIdx.x;          // 0..63
  const int mt   = blockIdx.y;
  const int lane = threadIdx.x;
  const int m     = mt * 16 + (lane & 15);
  const int khalf = lane >> 4;
  unsigned int* o = dst + (((size_t)mt * KT_ + kt) * 32 + lane) * 8;
#pragma unroll
  for (int j = 0; j < 8; ++j) {
    const int k = kt * 32 + (j & 3) * 2 + khalf * 8 + (j >> 2) * 16;
    const float f0 = X[(size_t)m * rowStride + k];
    const float f1 = X[(size_t)m * rowStride + k + 1];
    o[j] = (unsigned int)f2bf(f0) | ((unsigned int)f2bf(f1) << 16);
  }
}

// ---------------------------------------------------------------------------
// Fused attention-score kernel (dominant GEMM, 123 GFLOP):
//   scores[row,g] = sum_d tanh(tanh(A*Wv + bv)[row,d] * xq_att[b(row),d]) * Wa[d,g] + ba[g]
// Block: 256 threads (8 waves), 64 rows (4 m-tiles), all N (75 tiles).
// A (64 x 2048 bf16) staged in LDS with 1028-dword row pitch (bank-conflict free).
// Wave w handles n-tiles w, w+8, ...; 4 WMMA (one per m-tile) per B-tile load,
// so Wv bf16 L2 traffic = 392 blocks x 4.9 MB ~= 1.9 GB (L2-resident weights).
// ---------------------------------------------------------------------------
#define A_PITCH_U32 1028
#define SM_A_BYTES  (64 * A_PITCH_U32 * 4)             // 263168
#define SM_XQ_OFF   SM_A_BYTES
#define SM_XQ_BYTES (2 * DA_ * 4)                      // 9600
#define SM_PART_OFF (SM_XQ_OFF + SM_XQ_BYTES)
#define SM_PART_BYTES (8 * 64 * 4 * 4)                 // 8192
#define SM_TOTAL    (SM_PART_OFF + SM_PART_BYTES)      // 280960

__launch_bounds__(256, 1)
__global__ void att_scores_kernel(const float* __restrict__ input_v,
                                  const unsigned int* __restrict__ Wswz,
                                  const float* __restrict__ xq_att,
                                  const float* __restrict__ bv_att,
                                  const float* __restrict__ Wa,
                                  const float* __restrict__ ba,
                                  float* __restrict__ scores) {
  extern __shared__ unsigned char smem[];
  unsigned int* Alds   = (unsigned int*)smem;
  float*        xq_lds = (float*)(smem + SM_XQ_OFF);
  float*        part   = (float*)(smem + SM_PART_OFF);   // [wave][row64][g]

  const int tid   = threadIdx.x;
  const int mbase = blockIdx.x * 64;

  // ---- stage A: 64 rows x 2048 k, f32 -> bf16 (transpose-gather) ----
  {
    const int m   = tid & 63;
    const int row = mbase + m;
    const int bb  = row / S_;
    const int ss  = row - bb * S_;
    const float* src = input_v + (size_t)bb * (DV_ * S_) + ss;
    const int klane  = tid >> 6;                   // 0..3
    unsigned short* Ash = (unsigned short*)Alds;
#pragma unroll 4
    for (int it = 0; it < 512; ++it) {
      const int k = it * 4 + klane;
      Ash[m * (A_PITCH_U32 * 2) + k] = f2bf(src[(size_t)k * S_]);
    }
  }
  // ---- stage xq_att rows for the (<=2) batches this block touches ----
  const int b0 = mbase / S_;
  const int b1 = (mbase + 63) / S_;
  for (int i = tid; i < 2 * DA_; i += 256) {
    const int which = (i >= DA_) ? 1 : 0;
    xq_lds[i] = xq_att[(size_t)(which ? b1 : b0) * DA_ + (i - which * DA_)];
  }
  __syncthreads();

  const int wave  = tid >> 5;
  const int lane  = tid & 31;
  const int khalf = lane >> 4;
  const int nl    = lane & 15;
  const int thr   = (b0 + 1) * S_;    // rows >= thr belong to batch b1

  float sacc[4][8][4];
#pragma unroll
  for (int mt = 0; mt < 4; ++mt)
#pragma unroll
    for (int i = 0; i < 8; ++i)
#pragma unroll
      for (int g = 0; g < G_; ++g) sacc[mt][i][g] = 0.f;

  const float4* wa4 = (const float4*)Wa;

  for (int nt = wave; nt < NT_; nt += 8) {
    v8f acc[4];
    const v8f z = {0.f, 0.f, 0.f, 0.f, 0.f, 0.f, 0.f, 0.f};
#pragma unroll
    for (int mt = 0; mt < 4; ++mt) acc[mt] = z;

    const uint4v* bp =
        (const uint4v*)(Wswz + (size_t)nt * KT_ * 256 + (size_t)lane * 8);
#pragma unroll 2
    for (int kt = 0; kt < KT_; ++kt) {
      ABu bmat;
      bmat.q[0] = bp[0];
      bmat.q[1] = bp[1];
      bp += 64;                                   // advance 256 dwords
#pragma unroll
      for (int mt = 0; mt < 4; ++mt) {
        const uint4v* lp = (const uint4v*)(Alds + (size_t)(mt * 16 + nl) * A_PITCH_U32 +
                                           kt * 16 + khalf * 4);
        ABu a;
        a.q[0] = lp[0];
        a.q[1] = lp[2];                           // +8 dwords
        acc[mt] = wmma_bf16(a, bmat, acc[mt]);
      }
    }
    // fused epilogue: tanh -> *xq -> tanh -> *Wa, accumulate over d
    const int d   = nt * 16 + nl;
    const float bv = bv_att[d];
    const float4 w = wa4[d];
    const float xq0 = xq_lds[d];
    const float xq1 = xq_lds[DA_ + d];
#pragma unroll
    for (int mt = 0; mt < 4; ++mt) {
#pragma unroll
      for (int i = 0; i < 8; ++i) {
        const int row = mbase + mt * 16 + khalf * 8 + i;
        float xa = fast_tanh(acc[mt][i] + bv);
        xa = fast_tanh(xa * ((row >= thr) ? xq1 : xq0));
        sacc[mt][i][0] += xa * w.x;
        sacc[mt][i][1] += xa * w.y;
        sacc[mt][i][2] += xa * w.z;
        sacc[mt][i][3] += xa * w.w;
      }
    }
  }

  // butterfly-reduce the 16 d-lanes within each half (deterministic)
#pragma unroll
  for (int off = 8; off >= 1; off >>= 1) {
#pragma unroll
    for (int mt = 0; mt < 4; ++mt)
#pragma unroll
      for (int i = 0; i < 8; ++i)
#pragma unroll
        for (int g = 0; g < G_; ++g)
          sacc[mt][i][g] += __shfl_xor(sacc[mt][i][g], off);
  }
  if (nl == 0) {
#pragma unroll
    for (int mt = 0; mt < 4; ++mt)
#pragma unroll
      for (int i = 0; i < 8; ++i)
#pragma unroll
        for (int g = 0; g < G_; ++g)
          part[(size_t)wave * 256 + (mt * 16 + khalf * 8 + i) * 4 + g] = sacc[mt][i][g];
  }
  __syncthreads();
  // final cross-wave tree sum: 256 threads -> 64 rows x 4 glimpses
  {
    const int r = tid >> 2, g = tid & 3;
    float s = 0.f;
#pragma unroll
    for (int w = 0; w < 8; ++w) s += part[(size_t)w * 256 + r * 4 + g];
    scores[(size_t)(mbase + r) * G_ + g] = s + ba[g];
  }
}

// ---------------------------------------------------------------------------
// Softmax over the spatial axis per (b, g), in place on scores [B,S,G]
// ---------------------------------------------------------------------------
__global__ void softmax_kernel(float* __restrict__ scores) {
  __shared__ float red[256];
  const int tid = threadIdx.x;
  const int b = blockIdx.x >> 2, g = blockIdx.x & 3;
  float v = -1e30f;
  if (tid < S_) v = scores[(size_t)b * (S_ * G_) + tid * G_ + g];
  red[tid] = v;
  __syncthreads();
  for (int off = 128; off > 0; off >>= 1) {
    if (tid < off) red[tid] = fmaxf(red[tid], red[tid + off]);
    __syncthreads();
  }
  const float mx = red[0];
  __syncthreads();
  const float e = (tid < S_) ? __expf(v - mx) : 0.f;
  red[tid] = e;
  __syncthreads();
  for (int off = 128; off > 0; off >>= 1) {
    if (tid < off) red[tid] += red[tid + off];
    __syncthreads();
  }
  const float inv = 1.f / red[0];
  if (tid < S_) scores[(size_t)b * (S_ * G_) + tid * G_ + g] = e * inv;
}

// ---------------------------------------------------------------------------
// Glimpse pooling: v_att[b,g,c] = sum_s att[b,s,g] * v[b,c,s]
// grid (DV/256, B), block 256; float4 streaming over s (196 = 49*4)
// ---------------------------------------------------------------------------
__global__ void vatt_kernel(const float* __restrict__ input_v,
                            const float* __restrict__ att,
                            float* __restrict__ v_att) {
  __shared__ float att_s[S_ * G_];
  const int tid = threadIdx.x;
  const int b = blockIdx.y;
  for (int i = tid; i < S_ * G_; i += 256) att_s[i] = att[(size_t)b * (S_ * G_) + i];
  __syncthreads();
  const int c = blockIdx.x * 256 + tid;
  const float4* vp = (const float4*)(input_v + ((size_t)b * DV_ + c) * S_);
  float a0 = 0.f, a1 = 0.f, a2 = 0.f, a3 = 0.f;
#pragma unroll 7
  for (int s4 = 0; s4 < 49; ++s4) {
    const float4 qd = vp[s4];
    const int sb = s4 * 4 * G_;
    a0 += qd.x * att_s[sb + 0] + qd.y * att_s[sb + 4] + qd.z * att_s[sb + 8]  + qd.w * att_s[sb + 12];
    a1 += qd.x * att_s[sb + 1] + qd.y * att_s[sb + 5] + qd.z * att_s[sb + 9]  + qd.w * att_s[sb + 13];
    a2 += qd.x * att_s[sb + 2] + qd.y * att_s[sb + 6] + qd.z * att_s[sb + 10] + qd.w * att_s[sb + 14];
    a3 += qd.x * att_s[sb + 3] + qd.y * att_s[sb + 7] + qd.z * att_s[sb + 11] + qd.w * att_s[sb + 15];
  }
  v_att[((size_t)b * G_ + 0) * DV_ + c] = a0;
  v_att[((size_t)b * G_ + 1) * DV_ + c] = a1;
  v_att[((size_t)b * G_ + 2) * DV_ + c] = a2;
  v_att[((size_t)b * G_ + 3) * DV_ + c] = a3;
}

// ---------------------------------------------------------------------------
// Generic register-only WMMA GEMM (K=2048) on swizzled bf16 operands.
// mode 0: out = C + bias
// mode 1: out = tanh(C + bias)
// mode 2: out = tanh(tanh(C + bias) * emul[row,col])     (MLB fusion)
// grid (mtiles, ceil(ntiles/4)); block 128 (4 waves, 1 n-tile each)
// ---------------------------------------------------------------------------
__launch_bounds__(128)
__global__ void wmma_gemm_kernel(const unsigned int* __restrict__ Aswz,
                                 const unsigned int* __restrict__ Bswz,
                                 const float* __restrict__ bias,
                                 const float* __restrict__ emul,
                                 float* __restrict__ out,
                                 int n_tiles, int N,
                                 int out_rs, int out_co,
                                 int e_rs, int e_co, int mode) {
  const int wave = threadIdx.x >> 5;
  const int lane = threadIdx.x & 31;
  const int nt = blockIdx.y * 4 + wave;
  if (nt >= n_tiles) return;                       // wave-uniform
  const int mt = blockIdx.x;

  const v8f z = {0.f, 0.f, 0.f, 0.f, 0.f, 0.f, 0.f, 0.f};
  v8f c = z;
  const uint4v* ap = (const uint4v*)(Aswz + (size_t)mt * KT_ * 256 + (size_t)lane * 8);
  const uint4v* bp = (const uint4v*)(Bswz + (size_t)nt * KT_ * 256 + (size_t)lane * 8);
#pragma unroll 4
  for (int kt = 0; kt < KT_; ++kt) {
    ABu a, b;
    a.q[0] = ap[0]; a.q[1] = ap[1]; ap += 64;
    b.q[0] = bp[0]; b.q[1] = bp[1]; bp += 64;
    c = wmma_bf16(a, b, c);
  }
  const int col = nt * 16 + (lane & 15);
  if (col < N) {
    const int rbase = mt * 16 + ((lane >> 4) << 3);
#pragma unroll
    for (int i = 0; i < 8; ++i) {
      const int row = rbase + i;
      float v = c[i] + bias[col];
      if (mode >= 1) v = fast_tanh(v);
      if (mode == 2) v = fast_tanh(v * emul[(size_t)row * e_rs + e_co + col]);
      out[(size_t)row * out_rs + out_co + col] = v;
    }
  }
}

// ---------------------------------------------------------------------------
// Host-side launch
// ---------------------------------------------------------------------------
extern "C" void kernel_launch(void* const* d_in, const int* in_sizes, int n_in,
                              void* d_out, int out_size, void* d_ws, size_t ws_size,
                              hipStream_t stream) {
  const float* input_v = (const float*)d_in[0];
  const float* x_q     = (const float*)d_in[1];
  const float* Wv_att  = (const float*)d_in[2];
  const float* bv_att  = (const float*)d_in[3];
  const float* Wq_att  = (const float*)d_in[4];
  const float* bq_att  = (const float*)d_in[5];
  const float* Wa      = (const float*)d_in[6];
  const float* ba      = (const float*)d_in[7];
  const float* Wf      = (const float*)d_in[8];
  const float* bf      = (const float*)d_in[9];
  const float* Wqf     = (const float*)d_in[10];
  const float* bqf     = (const float*)d_in[11];
  const float* Wc      = (const float*)d_in[12];
  const float* bc      = (const float*)d_in[13];
  float* out = (float*)d_out;

  // --- workspace carve (all sizes are multiples of 256 B) ---
  unsigned char* ws = (unsigned char*)d_ws;
  size_t off = 0;
  auto carve = [&](size_t bytes) { unsigned char* p = ws + off; off += bytes; return p; };
  const size_t TILE_B = (size_t)KT_ * 256 * 4;           // bytes per 16-col/row strip
  unsigned int* wsWv  = (unsigned int*)carve(75  * TILE_B);
  unsigned int* wsWq  = (unsigned int*)carve(75  * TILE_B);
  unsigned int* wsWqf = (unsigned int*)carve(128 * TILE_B);
  unsigned int* wsWc  = (unsigned int*)carve(188 * TILE_B);
  unsigned int* wsWf  = (unsigned int*)carve(4 * 32 * TILE_B);
  unsigned int* wsAq  = (unsigned int*)carve(8 * TILE_B);
  unsigned int* wsAv  = (unsigned int*)carve(4 * 8 * TILE_B);
  unsigned int* wsAx  = (unsigned int*)carve(8 * TILE_B);
  float* ws_xqatt  = (float*)carve((size_t)B_ * DA_ * 4);
  float* ws_scores = (float*)carve((size_t)B_ * S_ * G_ * 4);
  float* ws_vatt   = (float*)carve((size_t)B_ * G_ * DV_ * 4);
  float* ws_xqf    = (float*)carve((size_t)B_ * DH_ * 4);
  float* ws_x      = (float*)carve((size_t)B_ * DH_ * 4);
  (void)in_sizes; (void)n_in; (void)out_size; (void)ws_size;

  const size_t WF_STRIDE_U = 32 * (size_t)KT_ * 256;     // dwords per glimpse weight
  const size_t AV_STRIDE_U = 8  * (size_t)KT_ * 256;     // dwords per glimpse A

  // 1) weight/activation conversion + swizzle (bf16, WMMA operand order)
  conv_swz_w<<<dim3(64, 75),  32, 0, stream>>>(Wv_att, wsWv, DA_, DA_);
  conv_swz_w<<<dim3(64, 75),  32, 0, stream>>>(Wq_att, wsWq, DA_, DA_);
  conv_swz_w<<<dim3(64, 128), 32, 0, stream>>>(Wqf, wsWqf, DH_, DH_);
  conv_swz_w<<<dim3(64, 188), 32, 0, stream>>>(Wc, wsWc, NANS_, NANS_);
  for (int g = 0; g < G_; ++g)
    conv_swz_w<<<dim3(64, 32), 32, 0, stream>>>(Wf + (size_t)g * DV_ * 512,
                                                wsWf + (size_t)g * WF_STRIDE_U, 512, 512);
  conv_swz_a<<<dim3(64, 8), 32, 0, stream>>>(x_q, wsAq, DV_);   // DQ == DV == 2048

  // 2) xq_att = tanh(x_q @ Wq_att + bq_att)
  wmma_gemm_kernel<<<dim3(8, 19), 128, 0, stream>>>(
      wsAq, wsWq, bq_att, nullptr, ws_xqatt, 75, DA_, DA_, 0, 0, 0, 1);

  // 3) fused big GEMM + MLB attention + score projection
  att_scores_kernel<<<dim3(MROWS_TOTAL / 64), 256, SM_TOTAL, stream>>>(
      input_v, wsWv, ws_xqatt, bv_att, Wa, ba, ws_scores);

  // 4) softmax over spatial, in place
  softmax_kernel<<<dim3(B_ * G_), 256, 0, stream>>>(ws_scores);

  // 5) glimpse pooling
  vatt_kernel<<<dim3(DV_ / 256, B_), 256, 0, stream>>>(input_v, ws_scores, ws_vatt);

  // 6) xqf = tanh(x_q @ Wqf + bqf)
  wmma_gemm_kernel<<<dim3(8, 32), 128, 0, stream>>>(
      wsAq, wsWqf, bqf, nullptr, ws_xqf, 128, DH_, DH_, 0, 0, 0, 1);

  // 7) per-glimpse fusion: x[:, g*512:] = tanh(tanh(v_att_g @ Wf_g + bf_g) * xqf)
  for (int g = 0; g < G_; ++g) {
    conv_swz_a<<<dim3(64, 8), 32, 0, stream>>>(ws_vatt + (size_t)g * DV_,
                                               wsAv + (size_t)g * AV_STRIDE_U, G_ * DV_);
    wmma_gemm_kernel<<<dim3(8, 8), 128, 0, stream>>>(
        wsAv + (size_t)g * AV_STRIDE_U, wsWf + (size_t)g * WF_STRIDE_U,
        bf + (size_t)g * 512, ws_xqf, ws_x,
        32, 512, DH_, g * 512, DH_, g * 512, 2);
  }

  // 8) classifier: out = x @ Wc + bc
  conv_swz_a<<<dim3(64, 8), 32, 0, stream>>>(ws_x, wsAx, DH_);
  wmma_gemm_kernel<<<dim3(8, 47), 128, 0, stream>>>(
      wsAx, wsWc, bc, nullptr, out, 188, NANS_, NANS_, 0, 0, 0, 0);
}
